// AttentionBlock_12962211299458
// MI455X (gfx1250) — matrix operand order repaired and checked by reading, not compiled, not run on hardware
//
#include <hip/hip_runtime.h>

typedef _Float16 h16;
typedef __attribute__((ext_vector_type(16))) _Float16 v16h;
typedef __attribute__((ext_vector_type(8)))  float    v8f;

#define B_   16
#define C_   512
#define T_   1024
#define NH_  8      // heads per batch
#define CH_  64     // channels per head
#define BH_  (B_ * NH_)

// ---------------- WMMA helpers (gfx1250 16x16x32 f16 -> f32) ----------------

__device__ __forceinline__ v8f wmma16(v16h a, v16h b, v8f c) {
  return __builtin_amdgcn_wmma_f32_16x16x32_f16(
      /*neg_a=*/false, a, /*neg_b=*/false, b,
      /*c_mod=*/(short)0, c, /*reuse_a=*/false, /*reuse_b=*/false);
}

// A fragment: 16x32 (MxK) f16 from row-major [M][K] source, stride lda.
// lane L: m = L%16, hi = L/16 ; elem e<8 -> K = hi*8+e ; e>=8 -> K = 16+hi*8+(e-8)
__device__ __forceinline__ v16h load_a_frag(const h16* base, int lda) {
  int lane = threadIdx.x & 31;
  int m = lane & 15, hi = lane >> 4;
  const h16* row = base + (size_t)m * lda + hi * 8;
  v16h a;
#pragma unroll
  for (int e = 0; e < 8; ++e) a[e] = row[e];
#pragma unroll
  for (int e = 0; e < 8; ++e) a[e + 8] = row[16 + e];
  return a;
}

// B fragment: 32x16 (KxN) f16 from TRANSPOSED [N][K] source, row stride ldn.
// lane L: n = L%16, hi = L/16 ; elem e<8 -> K = hi*8+e ; e>=8 -> K = 16+hi*8+(e-8)
__device__ __forceinline__ v16h load_bt_frag(const h16* base, int ldn) {
  int lane = threadIdx.x & 31;
  int n = lane & 15, hi = lane >> 4;
  const h16* p = base + (size_t)n * ldn + hi * 8;
  v16h b;
#pragma unroll
  for (int e = 0; e < 8; ++e) { b[e] = p[e]; b[e + 8] = p[16 + e]; }
  return b;
}

// ---------------- kernel 0: f32 -> f16 weight convert ----------------

__global__ void cvt_f16_kernel(const float* __restrict__ src, h16* __restrict__ dst, int n) {
  int i = blockIdx.x * blockDim.x + threadIdx.x;
  if (i < n) dst[i] = (h16)src[i];
}

// ---------------- kernel 1: GroupNorm -> f16, TRANSPOSED output ----------------
// one block per (b, group); output xn_t is [B, T, C] (t-major, C contiguous)

__global__ void gn_kernel(const float* __restrict__ x, const float* __restrict__ w,
                          const float* __restrict__ bb, h16* __restrict__ xn_t) {
  const int CG = 16;
  int b = blockIdx.x >> 5;
  int g = blockIdx.x & 31;
  const float* xg = x + ((size_t)b * C_ + (size_t)g * CG) * T_;

  float s = 0.f, s2 = 0.f;
  for (int i = threadIdx.x; i < CG * T_; i += 256) {
    float v = xg[i];
    s += v; s2 += v * v;
  }
  __shared__ float sh0[256], sh1[256];
  sh0[threadIdx.x] = s; sh1[threadIdx.x] = s2;
  __syncthreads();
  for (int off = 128; off > 0; off >>= 1) {
    if ((int)threadIdx.x < off) {
      sh0[threadIdx.x] += sh0[threadIdx.x + off];
      sh1[threadIdx.x] += sh1[threadIdx.x + off];
    }
    __syncthreads();
  }
  float invN = 1.0f / (CG * T_);
  float mean = sh0[0] * invN;
  float var  = sh1[0] * invN - mean * mean;
  float inv  = rsqrtf(var + 1e-5f);

  float wr[CG], br[CG];
#pragma unroll
  for (int cc = 0; cc < CG; ++cc) { wr[cc] = w[g * CG + cc] * inv; br[cc] = bb[g * CG + cc]; }

  // transpose-write: thread t writes 16 contiguous halves (32B run) per row.
  // reads are lane-coalesced (consecutive t, fixed c).
  h16* xo = xn_t + (size_t)b * T_ * C_ + (size_t)g * CG;
  for (int t = threadIdx.x; t < T_; t += 256) {
    h16 hv[CG];
#pragma unroll
    for (int cc = 0; cc < CG; ++cc)
      hv[cc] = (h16)((xg[(size_t)cc * T_ + t] - mean) * wr[cc] + br[cc]);
    h16* o = xo + (size_t)t * C_;
#pragma unroll
    for (int cc = 0; cc < CG; ++cc) o[cc] = hv[cc];
  }
}

// ---------------- kernel 2: QKV GEMM ----------------
// out[o,t] = sum_c W[o,c]*xn[c,t] + bias[o].  A = W [O][C], B = xn_t [T][C].
// One wave computes a 16(o) x 64(t) strip. Writes:
//   qt [bh, T, 64]  (t-major, * scale)  -> B operand of scores
//   kt [bh, T, 64]  (t-major, * scale)  -> A operand of scores
//   vc [bh, 64, T]  (c-major)           -> B operand of P*V

__global__ void qkv_kernel(const h16* __restrict__ W, const float* __restrict__ bias,
                           const h16* __restrict__ xn_t,
                           h16* __restrict__ qt, h16* __restrict__ kt, h16* __restrict__ vc) {
  const float SCALE = 0.35355339059327373f; // 64^-0.25
  int wid = (int)((blockIdx.x * blockDim.x + threadIdx.x) >> 5);
  int ot = wid % 96;              // o tile (3C/16 = 96)
  int ts = (wid / 96) & 15;       // t strip of 64
  int b  = wid / (96 * 16);
  int o0 = ot * 16, t0 = ts * 64;
  int lane = threadIdx.x & 31, nn = lane & 15, hi = lane >> 4;

  v8f acc[4] = {};
  const h16* xb = xn_t + (size_t)b * T_ * C_;
  for (int k0 = 0; k0 < C_; k0 += 32) {
    const h16* wp = W + (size_t)o0 * C_ + k0;
    __builtin_prefetch(wp + 32, 0, 1);
    v16h a = load_a_frag(wp, C_);
#pragma unroll
    for (int j = 0; j < 4; ++j) {
      const h16* bp = xb + (size_t)(t0 + j * 16) * C_ + k0;
      __builtin_prefetch(bp + 32, 0, 1);
      v16h bf = load_bt_frag(bp, C_);
      acc[j] = wmma16(a, bf, acc[j]);
    }
  }

  float myb = bias[o0 + nn];
  int seg  = o0 >> 9;             // 0=q 1=k 2=v (uniform per tile)
  int oo   = o0 & 511;
  int head = oo >> 6;
  int c0   = oo & 63;             // multiple of 16
  size_t bh = (size_t)b * NH_ + head;

  if (seg == 0) {
    h16* dst = qt + bh * T_ * CH_;
#pragma unroll
    for (int j = 0; j < 4; ++j) {
      int t = t0 + j * 16 + nn;
#pragma unroll
      for (int r = 0; r < 8; ++r) {
        int m = r + hi * 8;
        float bv = __shfl(myb, m, 32);
        dst[(size_t)t * CH_ + c0 + m] = (h16)((acc[j][r] + bv) * SCALE);
      }
    }
  } else if (seg == 1) {
    h16* dst = kt + bh * T_ * CH_;
#pragma unroll
    for (int j = 0; j < 4; ++j) {
      int t = t0 + j * 16 + nn;
#pragma unroll
      for (int r = 0; r < 8; ++r) {
        int m = r + hi * 8;
        float bv = __shfl(myb, m, 32);
        dst[(size_t)t * CH_ + c0 + m] = (h16)((acc[j][r] + bv) * SCALE);
      }
    }
  } else {
    h16* dst = vc + bh * CH_ * T_;
#pragma unroll
    for (int j = 0; j < 4; ++j) {
      int t = t0 + j * 16 + nn;
#pragma unroll
      for (int r = 0; r < 8; ++r) {
        int m = r + hi * 8;
        float bv = __shfl(myb, m, 32);
        dst[(size_t)(c0 + m) * T_ + t] = (h16)(acc[j][r] + bv);
      }
    }
  }
}

// ---------------- kernel 3: attention (flash-style online softmax) ----------------
// One wave per (bh, 16-row t-tile). Scores computed TRANSPOSED (s rows, t lanes)
// so the exp()'d tile is directly the A fragment of the P*V WMMA.
// a_out is t-major [B, T, C] -> B operand of proj.

__global__ void attn_kernel(const h16* __restrict__ qt, const h16* __restrict__ kt,
                            const h16* __restrict__ vc, h16* __restrict__ a_out) {
  int wid = (int)((blockIdx.x * blockDim.x + threadIdx.x) >> 5);
  int tt = wid & 63;
  int bh = wid >> 6;
  int t0 = tt * 16;
  int lane = threadIdx.x & 31, nn = lane & 15, hi = lane >> 4;

  const h16* qb = qt + (size_t)bh * T_ * CH_;
  const h16* kb = kt + (size_t)bh * T_ * CH_;
  const h16* vb = vc + (size_t)bh * CH_ * T_;

  // q B-fragments (N=t rows, K=c contiguous): invariant over the s loop
  v16h qf0 = load_bt_frag(qb + (size_t)t0 * CH_, CH_);       // c 0..31
  v16h qf1 = load_bt_frag(qb + (size_t)t0 * CH_ + 32, CH_);  // c 32..63

  float mrow = -1e30f, lrow = 0.f;
  v8f O[4] = {};

  for (int s0 = 0; s0 < T_; s0 += 32) {
    __builtin_prefetch(kb + (size_t)(s0 + 32) * CH_, 0, 1);
    // score_T[s, t] = sum_c kt[s,c] * q[c,t] ; two 16-s tiles
    v8f S0 = {}, S1 = {};
    v16h ka0 = load_a_frag(kb + (size_t)s0 * CH_, CH_);
    v16h ka1 = load_a_frag(kb + (size_t)s0 * CH_ + 32, CH_);
    S0 = wmma16(ka0, qf0, S0);
    S0 = wmma16(ka1, qf1, S0);
    v16h kc0 = load_a_frag(kb + (size_t)(s0 + 16) * CH_, CH_);
    v16h kc1 = load_a_frag(kb + (size_t)(s0 + 16) * CH_ + 32, CH_);
    S1 = wmma16(kc0, qf0, S1);
    S1 = wmma16(kc1, qf1, S1);

    // online softmax over s, keyed per-lane by t (lanes L and L^16 hold
    // complementary s-halves of the same t column)
    float cmax = -1e30f;
#pragma unroll
    for (int r = 0; r < 8; ++r) { cmax = fmaxf(cmax, S0[r]); cmax = fmaxf(cmax, S1[r]); }
    cmax = fmaxf(cmax, __shfl_xor(cmax, 16, 32));
    float mnew = fmaxf(mrow, cmax);
    float corr = __expf(mrow - mnew);

    float psum = 0.f;
    v16h wfrag;
#pragma unroll
    for (int r = 0; r < 8; ++r) {
      float p0 = __expf(S0[r] - mnew);
      float p1 = __expf(S1[r] - mnew);
      psum += p0 + p1;
      wfrag[r]     = (h16)p0;   // A-frag elem e=r   : s_rel = r + hi*8
      wfrag[r + 8] = (h16)p1;   // A-frag elem e=r+8 : s_rel = 16 + r + hi*8
    }
    psum += __shfl_xor(psum, 16, 32);
    lrow = lrow * corr + psum;
    mrow = mnew;

    // rescale O: row m = r + hi*8 of O corresponds to t held in lane (r + hi*8)
    float corrB[8];
#pragma unroll
    for (int r = 0; r < 8; ++r) corrB[r] = __shfl(corr, r + hi * 8, 32);
#pragma unroll
    for (int j = 0; j < 4; ++j)
#pragma unroll
      for (int r = 0; r < 8; ++r) O[j][r] *= corrB[r];

    // O[t, c] += wfrag(16t x 32s) * v(32s x 16c); B from c-major v: [c][s]
#pragma unroll
    for (int j = 0; j < 4; ++j) {
      v16h vf = load_bt_frag(vb + (size_t)(j * 16) * T_ + s0, T_);
      O[j] = wmma16(wfrag, vf, O[j]);
    }
  }

  float linv = 1.0f / lrow;
  float linvB[8];
#pragma unroll
  for (int r = 0; r < 8; ++r) linvB[r] = __shfl(linv, r + hi * 8, 32);

  int b = bh >> 3, head = bh & 7;
  h16* ab = a_out + (size_t)b * T_ * C_ + (size_t)head * CH_;
#pragma unroll
  for (int j = 0; j < 4; ++j)
#pragma unroll
    for (int r = 0; r < 8; ++r) {
      int c = j * 16 + nn;
      int t = t0 + r + hi * 8;
      ab[(size_t)t * C_ + c] = (h16)(O[j][r] * linvB[r]);
    }
}

// ---------------- kernel 4: proj GEMM + bias + residual ----------------
// A = proj_w [O][C], B = a_t [T][C] (transposed layout -> contiguous frags)

__global__ void proj_kernel(const h16* __restrict__ W, const float* __restrict__ bias,
                            const h16* __restrict__ a_t, const float* __restrict__ x,
                            float* __restrict__ out) {
  int wid = (int)((blockIdx.x * blockDim.x + threadIdx.x) >> 5);
  int ot = wid & 31;             // C/16 = 32
  int ts = (wid >> 5) & 15;
  int b  = wid >> 9;
  int o0 = ot * 16, t0 = ts * 64;
  int lane = threadIdx.x & 31, nn = lane & 15, hi = lane >> 4;

  v8f acc[4] = {};
  const h16* ab = a_t + (size_t)b * T_ * C_;
  for (int k0 = 0; k0 < C_; k0 += 32) {
    const h16* wp = W + (size_t)o0 * C_ + k0;
    __builtin_prefetch(wp + 32, 0, 1);
    v16h af = load_a_frag(wp, C_);
#pragma unroll
    for (int j = 0; j < 4; ++j) {
      const h16* bp = ab + (size_t)(t0 + j * 16) * C_ + k0;
      __builtin_prefetch(bp + 32, 0, 1);
      v16h bf = load_bt_frag(bp, C_);
      acc[j] = wmma16(af, bf, acc[j]);
    }
  }

  float myb = bias[o0 + nn];
  const float* xb = x   + ((size_t)b * C_ + o0) * T_;
  float*       ob = out + ((size_t)b * C_ + o0) * T_;
#pragma unroll
  for (int r = 0; r < 8; ++r) {
    int m = r + hi * 8;
    float bv = __shfl(myb, m, 32);
#pragma unroll
    for (int j = 0; j < 4; ++j) {
      int t = t0 + j * 16 + nn;
      ob[(size_t)m * T_ + t] = xb[(size_t)m * T_ + t] + acc[j][r] + bv;
    }
  }
}

// ---------------- launch ----------------

extern "C" void kernel_launch(void* const* d_in, const int* in_sizes, int n_in,
                              void* d_out, int out_size, void* d_ws, size_t ws_size,
                              hipStream_t stream) {
  const float* x      = (const float*)d_in[0];
  const float* norm_w = (const float*)d_in[1];
  const float* norm_b = (const float*)d_in[2];
  const float* qkv_w  = (const float*)d_in[3];
  const float* qkv_b  = (const float*)d_in[4];
  const float* proj_w = (const float*)d_in[5];
  const float* proj_b = (const float*)d_in[6];
  float* out = (float*)d_out;

  char* ws = (char*)d_ws;
  size_t off = 0;
  auto alloc = [&](size_t bytes) -> void* {
    off = (off + 255) & ~(size_t)255;
    void* p = ws + off;
    off += bytes;
    return p;
  };

  h16* xnt = (h16*)alloc((size_t)B_ * T_ * C_ * 2);      // 16 MB; reused as a_t
  h16* qwf = (h16*)alloc((size_t)3 * C_ * C_ * 2);       // 1.5 MB
  h16* pwf = (h16*)alloc((size_t)C_ * C_ * 2);           // 0.5 MB
  h16* qtb = (h16*)alloc((size_t)BH_ * T_ * CH_ * 2);    // 16 MB
  h16* ktb = (h16*)alloc((size_t)BH_ * T_ * CH_ * 2);    // 16 MB
  h16* vcb = (h16*)alloc((size_t)BH_ * CH_ * T_ * 2);    // 16 MB
  h16* abuf = xnt;   // xn_t dead after qkv_kernel; reuse for attention output

  {
    int n1 = 3 * C_ * C_;
    cvt_f16_kernel<<<(n1 + 255) / 256, 256, 0, stream>>>(qkv_w, qwf, n1);
    int n2 = C_ * C_;
    cvt_f16_kernel<<<(n2 + 255) / 256, 256, 0, stream>>>(proj_w, pwf, n2);
  }
  gn_kernel<<<B_ * 32, 256, 0, stream>>>(x, norm_w, norm_b, xnt);

  // qkv: 96 o-tiles * 16 t-strips * 16 batches waves, 8 waves/block
  qkv_kernel<<<(96 * 16 * B_) / 8, 256, 0, stream>>>(qwf, qkv_b, xnt, qtb, ktb, vcb);

  // attention: 128 bh * 64 t-tiles waves
  attn_kernel<<<(BH_ * 64) / 8, 256, 0, stream>>>(qtb, ktb, vcb, abuf);

  // proj + residual: 32 o-tiles * 16 t-strips * 16 batches waves
  proj_kernel<<<(32 * 16 * B_) / 8, 256, 0, stream>>>(pwf, proj_b, abuf, x, out);
}
